// PSSM_70179765616886
// MI455X (gfx1250) — matrix operand order repaired and checked by reading, not compiled, not run on hardware
//
#include <hip/hip_runtime.h>

#define B_SZ   2
#define SEQ    2048
#define DIM    64
#define DI     128      // d_inner
#define NST    64       // d_state
#define CHUNK  512
#define NCHUNK (SEQ / CHUNK)

typedef __attribute__((ext_vector_type(16))) __bf16        v16bf;
typedef __attribute__((ext_vector_type(8)))  float         v8f;
typedef __attribute__((ext_vector_type(4)))  unsigned int  v4u;
typedef __attribute__((ext_vector_type(8)))  int           v8i_t;
typedef __attribute__((ext_vector_type(4)))  int           v4i_t;

__device__ __forceinline__ __bf16 tobf(float f) {
    union { float f; unsigned u; } uf; uf.f = f;
    union { unsigned short s; __bf16 h; } o;
    o.s = (unsigned short)((uf.u + 0x8000u) >> 16);     // round-half-up: 1 VALU op
    return o.h;
}

__device__ __forceinline__ float fast_sigmoid(float v) {
    return __builtin_amdgcn_rcpf(1.0f + __expf(-v));    // v_rcp_f32, no IEEE divide
}

// ---------------------------------------------------------------------------
// TDM: 1-D contiguous copy global -> LDS via tensor_load_to_lds (ISA ch.8 D#).
// group0: count=1 | lds_addr | global_addr[56:0] | type=2
// group1: data_size=4B, tensor_dim0=tile_dim0=nelem, tensor_dim1=1, stride=nelem
// ---------------------------------------------------------------------------
__device__ __forceinline__ void tdm_load_1d(const float* gsrc, unsigned lds_byte, int nelem) {
    unsigned long long ga = (unsigned long long)gsrc;
    v4u g0;
    g0[0] = 1u;                                          // count=1 (valid user D#)
    g0[1] = lds_byte;                                    // lds_addr (bytes)
    g0[2] = (unsigned)ga;                                // global_addr[31:0]
    g0[3] = (unsigned)((ga >> 32) & 0x01FFFFFFull) | (2u << 30);  // addr[56:32] | type=2
    v8i_t g1;
    g1[0] = (int)(2u << 16);                             // data_size = 2 (4 bytes)
    g1[1] = (int)((unsigned)nelem << 16);                // tensor_dim0[15:0]
    g1[2] = (int)((((unsigned)nelem >> 16) & 0xFFFFu) | (1u << 16)); // td0 hi | tensor_dim1=1
    g1[3] = (int)((unsigned)nelem << 16);                // tile_dim0
    g1[4] = 0;                                           // tile_dim1=0 (1-D), tile_dim2=0
    g1[5] = nelem;                                       // tensor_dim0_stride lo
    g1[6] = 0;
    g1[7] = 0;
    v4i_t z4 = {0, 0, 0, 0};
#if __clang_major__ >= 23
    v8i_t z8 = {0, 0, 0, 0, 0, 0, 0, 0};
    __builtin_amdgcn_tensor_load_to_lds(g0, g1, z4, z4, z8, 0);
#else
    __builtin_amdgcn_tensor_load_to_lds(g0, g1, z4, z4, 0);
#endif
}

// ---------------------------------------------------------------------------
// Kernel 0: zero output, compute a[i] = sum_j A[i][j]
// ---------------------------------------------------------------------------
__global__ void pssm_init(float* __restrict__ out, const float* __restrict__ A,
                          float* __restrict__ a_vec) {
    int tid = blockIdx.x * 256 + threadIdx.x;            // grid covers 524288 exactly
    out[tid] = 0.0f;
    if (blockIdx.x == 0 && threadIdx.x < NST) {
        float s = 0.0f;
        #pragma unroll 8
        for (int j = 0; j < NST; ++j) s += A[threadIdx.x * NST + j];
        a_vec[threadIdx.x] = s;
    }
}

// ---------------------------------------------------------------------------
// Kernel 1: xp = x @ W^T via WMMA bf16 (f32 accumulate).
// One wave computes a 16(M) x 64(N) strip: A operand loaded once, 4 B tiles,
// 8 x v_wmma_f32_16x16x32_bf16. N-group (64 cols) selects one output region:
// ng 0/1 -> xc, 2/3 -> sigmoid -> delta_t[b][d][l], 4 -> Bmat, 5 -> Cmat.
// ---------------------------------------------------------------------------
__global__ __launch_bounds__(128) void pssm_proj(
        const float* __restrict__ x, const float* __restrict__ W,
        float* __restrict__ xc, float* __restrict__ delta_t,
        float* __restrict__ Bmat, float* __restrict__ Cmat) {
    const int wave = threadIdx.x >> 5;
    const int lane = threadIdx.x & 31;
    const int id   = blockIdx.x * 4 + wave;              // 0..1535
    const int ng   = id % 6;                             // N-group (64 cols)
    const int tm   = id / 6;                             // M tile (16 rows)

    const int row  = tm * 16 + (lane & 15);              // global M = b*SEQ + l
    const int kb_a = (lane & 16) ? 8  : 0;               // A operand K base per half-wave
    const int kb_b = (lane & 16) ? 16 : 0;               // unused: B loads index with kb_a, same k order as A
    const float* __restrict__ xrow = x + row * DIM;

    v16bf a0, a1;
    #pragma unroll
    for (int e = 0; e < 8; ++e) {                        // A 16x32 bf16 layout x2 (K=64)
        a0[e]     = tobf(xrow[kb_a + e]);
        a0[e + 8] = tobf(xrow[16 + kb_a + e]);
        a1[e]     = tobf(xrow[32 + kb_a + e]);
        a1[e + 8] = tobf(xrow[48 + kb_a + e]);
    }

    v8f acc[4];
    #pragma unroll
    for (int j = 0; j < 4; ++j) {
        const int ncol = ng * 64 + j * 16 + (lane & 15); // W row for this B tile
        const float* __restrict__ wrow = W + ncol * DIM;
        v16bf b0, b1;
        #pragma unroll
        for (int e = 0; e < 16; ++e) {                   // B 32x16 bf16 layout x2
            b0[e] = tobf(wrow[kb_a + e + (e < 8 ? 0 : 8)]);
            b1[e] = tobf(wrow[32 + kb_a + e + (e < 8 ? 0 : 8)]);
        }
        v8f c = {};
        c = __builtin_amdgcn_wmma_f32_16x16x32_bf16(false, a0, false, b0, (short)0, c, false, false);
        c = __builtin_amdgcn_wmma_f32_16x16x32_bf16(false, a1, false, b1, (short)0, c, false, false);
        acc[j] = c;
    }

    #pragma unroll
    for (int j = 0; j < 4; ++j) {
        const int colN = ng * 64 + j * 16 + (lane & 15);
        #pragma unroll
        for (int r = 0; r < 8; ++r) {
            const int m = tm * 16 + r + ((lane & 16) ? 8 : 0);  // D: lanes 16-31 -> M+8
            const float v = acc[j][r];
            if (ng < 2) {                                        // x_c_in
                xc[m * DI + colN] = v;
            } else if (ng < 4) {                                 // delta = sigmoid(raw)
                const int d = colN - DI;
                const int b = m >> 11, l = m & (SEQ - 1);
                delta_t[(b * DI + d) * SEQ + l] = fast_sigmoid(v);
            } else if (ng == 4) {                                // B_in
                Bmat[m * NST + (colN - 2 * DI)] = v;
            } else {                                             // C_in
                Cmat[m * NST + (colN - 2 * DI - NST)] = v;
            }
        }
    }
}

// ---------------------------------------------------------------------------
// Kernel 2: depthwise causal conv (width 4) + bias; u = delta * x_conv.
// ---------------------------------------------------------------------------
__global__ void pssm_conv(const float* __restrict__ xc, const float* __restrict__ delta_t,
                          const float* __restrict__ cw, const float* __restrict__ cb,
                          float* __restrict__ xconv_t, float* __restrict__ u_t) {
    const int tid  = blockIdx.x * 256 + threadIdx.x;     // 524288 total = [b][d][l]
    const int l    = tid & (SEQ - 1);
    const int rest = tid >> 11;
    const int d    = rest & (DI - 1);
    const int b    = rest >> 7;

    float acc = cb[d];
    #pragma unroll
    for (int j = 0; j < 4; ++j) {
        const int ll = l - 3 + j;
        if (ll >= 0) acc += cw[d * 4 + j] * xc[(b * SEQ + ll) * DI + d];
    }
    const float s = delta_t[tid];
    xconv_t[tid] = acc;
    u_t[tid]     = s * acc;
}

// ---------------------------------------------------------------------------
// Kernel 3: bidirectional selective scan + output contraction.
// Block = 2 waves (fwd / bwd), one block per (b,d). delta/u streamed into LDS
// by the Tensor Data Mover, double-buffered per wave, synced via TENSORcnt
// (in-order per wave: wait<=2 retires the previous chunk while the next two
// descriptors are in flight). B/C rows prefetched with global_prefetch_b8.
// ---------------------------------------------------------------------------
__global__ __launch_bounds__(64) void pssm_scan(
        const float* __restrict__ delta_t, const float* __restrict__ u_t,
        const float* __restrict__ xconv_t, const float* __restrict__ Bmat,
        const float* __restrict__ Cmat,    const float* __restrict__ a_vec,
        const float* __restrict__ Dv,      float* __restrict__ out) {
    __shared__ float sh[4096];                    // [wave][buf][{delta,u}][CHUNK] = 16KB
    const int bd   = blockIdx.x;                  // 0..255 = b*128 + d
    const int b    = bd >> 7;
    const int d    = bd & (DI - 1);
    const int wave = threadIdx.x >> 5;
    const int lane = threadIdx.x & 31;
    const bool fwd = (wave == 0);

    const float a0 = a_vec[lane];
    const float a1 = a_vec[lane + 32];
    const int  base = bd * SEQ;
    const float dcoef = Dv[d];
    const int  woff = wave * 2048;                // floats
    const unsigned shbyte = (unsigned)(unsigned long long)&sh[0];  // LDS offset = addr[31:0]

    // stage chunk c into buffer `buf` for this wave's direction
    auto issue = [&](int c, int buf) {
        const int lbeg = fwd ? c * CHUNK : (SEQ - (c + 1) * CHUNK);
        const unsigned ldsd = shbyte + (unsigned)(woff + buf * 1024) * 4u;
        tdm_load_1d(delta_t + base + lbeg, ldsd, CHUNK);
        tdm_load_1d(u_t     + base + lbeg, ldsd + CHUNK * 4u, CHUNK);
    };

    issue(0, 0);
    float h0 = 0.0f, h1 = 0.0f;
    for (int c = 0; c < NCHUNK; ++c) {
        const int buf = c & 1;
        if (c + 1 < NCHUNK) {
            issue(c + 1, buf ^ 1);
            __builtin_amdgcn_s_wait_tensorcnt(2);  // previous chunk's 2 xfers done
        } else {
            __builtin_amdgcn_s_wait_tensorcnt(0);
        }
        asm volatile("" ::: "memory");             // LDS now holds chunk c
        const int lbeg = fwd ? c * CHUNK : (SEQ - (c + 1) * CHUNK);
        const float* shd = &sh[woff + buf * 1024];
        const float* shu = shd + CHUNK;

        for (int s = 0; s < CHUNK; ++s) {
            const int idx = fwd ? s : (CHUNK - 1 - s);
            const int l   = lbeg + idx;
            const float dl = shd[idx];
            const float ul = shu[idx];
            const int bc = (b * SEQ + l) * NST;
            {
                const int lp = fwd ? (l + 8) : (l - 8);
                if (lp >= 0 && lp < SEQ) {
                    const int bp = (b * SEQ + lp) * NST;
                    __builtin_prefetch(&Bmat[bp + lane], 0, 0);
                    __builtin_prefetch(&Cmat[bp + lane], 0, 0);
                }
            }
            const float Bv0 = Bmat[bc + lane], Bv1 = Bmat[bc + lane + 32];
            const float Cv0 = Cmat[bc + lane], Cv1 = Cmat[bc + lane + 32];

            h0 = __expf(a0 * dl) * h0 + ul * Bv0;
            h1 = __expf(a1 * dl) * h1 + ul * Bv1;

            float p = h0 * Cv0 + h1 * Cv1;
            p += __shfl_xor(p, 16);
            p += __shfl_xor(p, 8);
            p += __shfl_xor(p, 4);
            p += __shfl_xor(p, 2);
            p += __shfl_xor(p, 1);
            if (lane == 0) {
                if (fwd) p += dcoef * xconv_t[base + l];
                atomicAdd(&out[(b * SEQ + l) * DI + d], p);
            }
        }
    }
}

// ---------------------------------------------------------------------------
extern "C" void kernel_launch(void* const* d_in, const int* in_sizes, int n_in,
                              void* d_out, int out_size, void* d_ws, size_t ws_size,
                              hipStream_t stream) {
    const float* x  = (const float*)d_in[0];   // (2,2048,64)
    const float* W  = (const float*)d_in[1];   // (384,64)
    const float* cw = (const float*)d_in[2];   // (128,1,4)
    const float* cb = (const float*)d_in[3];   // (128,)
    const float* A  = (const float*)d_in[4];   // (64,64)
    const float* Dv = (const float*)d_in[5];   // (128,)
    float* out = (float*)d_out;                // (2,2048,128) = 524288

    float* ws      = (float*)d_ws;
    float* xc      = ws;                       // 524288
    float* delta_t = xc      + 524288;         // 524288  [b][d][l]
    float* xconv_t = delta_t + 524288;         // 524288  [b][d][l]
    float* u_t     = xconv_t + 524288;         // 524288  [b][d][l]
    float* Bmat    = u_t     + 524288;         // 262144  [b][l][i]
    float* Cmat    = Bmat    + 262144;         // 262144  [b][l][i]
    float* a_vec   = Cmat    + 262144;         // 64

    pssm_init<<<2048, 256, 0, stream>>>(out, A, a_vec);
    pssm_proj<<<384, 128, 0, stream>>>(x, W, xc, delta_t, Bmat, Cmat);
    pssm_conv<<<2048, 256, 0, stream>>>(xc, delta_t, cw, cb, xconv_t, u_t);
    pssm_scan<<<256, 64, 0, stream>>>(delta_t, u_t, xconv_t, Bmat, Cmat, a_vec, Dv, out);
}
